// CDE_23175643529380
// MI455X (gfx1250) — matrix operand order repaired and checked by reading, not compiled, not run on hardware
//
#include <hip/hip_runtime.h>
#include <hip/hip_bf16.h>

typedef __bf16 bf16_t;
typedef bf16_t v16bf __attribute__((ext_vector_type(16)));
typedef bf16_t v8bf  __attribute__((ext_vector_type(8)));
typedef float  v8f   __attribute__((ext_vector_type(8)));

#define T_LEN 1024
#define NBATCH 256
#define HDIM 64
#define IDIM 8
#define ODIM 8
#define CDIM 10
#define ROWS 32              // batch rows per workgroup
#define NWG (NBATCH / ROWS)  // 8 workgroups

// ---- async global->LDS support (toolchain-dependent; guarded) ----
#if defined(__has_builtin)
#if __has_builtin(__builtin_amdgcn_global_load_async_to_lds_b32)
#define HAVE_ASYNC_LDS 1
#endif
#endif

typedef __attribute__((address_space(1))) int gint_as1;
typedef __attribute__((address_space(3))) int lint_as3;

__device__ __forceinline__ void wait_asynccnt0() {
#if defined(__has_builtin) && __has_builtin(__builtin_amdgcn_s_wait_asynccnt)
    __builtin_amdgcn_s_wait_asynccnt(0);
#else
    asm volatile("s_wait_asynccnt 0" ::: "memory");
#endif
}

// ---------------- dXdt precompute: dXw[s][b][i], s in [0, T-2] ----------------
__global__ void cde_dx_kernel(const float* __restrict__ coeffs,
                              const float* __restrict__ times,
                              float* __restrict__ dXw) {
    int idx = blockIdx.x * blockDim.x + threadIdx.x;
    const int total = (T_LEN - 1) * NBATCH * IDIM;
    if (idx >= total) return;
    int i = idx & 7;
    int b = (idx >> 3) & (NBATCH - 1);
    int s = idx >> 11;                       // / (256*8)
    int tt = (s == 0) ? 0 : (s - 1);
    float frac = (s == 0) ? 0.0f : (times[s] - times[s - 1]);
    const float* cb = coeffs + ((size_t)b * (T_LEN - 1) + tt) * 32;
    float bc = cb[8 + i], cc = cb[16 + i], dc = cb[24 + i];
    dXw[idx] = bc + (2.0f * cc + 3.0f * dc * frac) * frac;
}

// ---------------- WMMA fragment loaders (bf16, row-major LDS, ld=64) ----------
__device__ __forceinline__ v16bf load_frag_a(const bf16_t* src, int rowBase, int lane, int kt) {
    // A 16x32 bf16: lanes<16 -> K {kt+0..7, kt+16..23}; lanes>=16 -> K {kt+8..15, kt+24..31}
    int row  = rowBase + (lane & 15);
    int koff = kt + ((lane < 16) ? 0 : 8);
    union { v16bf v; struct { v8bf lo, hi; } p; } u;
    u.p.lo = *reinterpret_cast<const v8bf*>(src + row * 64 + koff);
    u.p.hi = *reinterpret_cast<const v8bf*>(src + row * 64 + koff + 16);
    return u.v;
}
__device__ __forceinline__ v16bf load_frag_b(const bf16_t* src, int nBase, int lane, int kt) {
    // B 32x16 bf16 from weight stored [N][K]: lanes<16 -> K {kt+0..7, kt+16..23}; lanes>=16 -> K {kt+8..15, kt+24..31}
    int n    = nBase + (lane & 15);
    int koff = kt + ((lane < 16) ? 0 : 8);
    union { v16bf v; struct { v8bf lo, hi; } p; } u;
    u.p.lo = *reinterpret_cast<const v8bf*>(src + n * 64 + koff);
    u.p.hi = *reinterpret_cast<const v8bf*>(src + n * 64 + koff + 16);
    return u.v;
}

__device__ __forceinline__ v8f gemm_pre(v16bf a0, v16bf a1, v16bf b0, v16bf b1, float biasv) {
    v8f acc;
#pragma unroll
    for (int i = 0; i < 8; ++i) acc[i] = biasv;
    acc = __builtin_amdgcn_wmma_f32_16x16x32_bf16(false, a0, false, b0, (short)0, acc, false, false);
    acc = __builtin_amdgcn_wmma_f32_16x16x32_bf16(false, a1, false, b1, (short)0, acc, false, false);
    return acc;
}

__device__ __forceinline__ void store_act(bf16_t* dst, v8f acc, int mt, int ntb,
                                          int lane, bool lipswish) {
    int col   = ntb + (lane & 15);
    int rbase = mt * 16 + ((lane >= 16) ? 8 : 0);
#pragma unroll
    for (int r = 0; r < 8; ++r) {
        float x = acc[r];
        if (lipswish) x = 0.909f * x / (1.0f + __expf(-x));
        dst[(rbase + r) * 64 + col] = (bf16_t)x;
    }
}

// ---------------- persistent scan kernel ----------------
__global__ __launch_bounds__(256, 1) void cde_scan_kernel(
    const float* __restrict__ coeffs, const float* __restrict__ times,
    const int* __restrict__ maskp,
    const float* __restrict__ Win,  const float* __restrict__ b_in,
    const float* __restrict__ Wf1,  const float* __restrict__ bf1,
    const float* __restrict__ Wf2,  const float* __restrict__ bf2,
    const float* __restrict__ Wout, const float* __restrict__ bout,
    const float* __restrict__ Wini, const float* __restrict__ bini,
    const float* __restrict__ Wdec, const float* __restrict__ bdec,
    const float* __restrict__ Wcls, const float* __restrict__ bcls,
    const float* __restrict__ dXw,
    float* __restrict__ outPred, float* __restrict__ outLog) {

    __shared__ bf16_t sWz[64 * 64];       // Win[:,1:]  (N x K)
    __shared__ bf16_t sWf1[64 * 64];
    __shared__ bf16_t sWf2[64 * 64];
    __shared__ bf16_t sWout[512 * 64];
    __shared__ bf16_t sWdecB[16 * 64];    // Wdec padded to 16 rows (bf16 B-matrix)
    __shared__ float  sZ[ROWS * 64];      // fp32 master z
    __shared__ bf16_t sA[ROWS * 64];      // activation ping
    __shared__ bf16_t sB[ROWS * 64];      // activation pong
    __shared__ float  sDX[8 * ROWS];      // [i][row]
    __shared__ float  sW0[64], sBin[64], sBf1[64], sBf2[64];
    __shared__ float  sBout[512];
    __shared__ float  sBdec[8];
    __shared__ float  sWcls[10 * 64];
    __shared__ float  sBcls[10];
    __shared__ int    sLen[ROWS];

    const int tid  = threadIdx.x;
    const int lane = tid & 31;
    const int w    = tid >> 5;
    const int wu   = __builtin_amdgcn_readfirstlane(w);  // wave-uniform scalar copy
    const int bb   = blockIdx.x * ROWS;

    // ---- stage weights to LDS ----
    for (int e = tid; e < 64 * 64; e += 256) {
        sWz[e]  = (bf16_t)Win[(e >> 6) * 65 + 1 + (e & 63)];
        sWf1[e] = (bf16_t)Wf1[e];
        sWf2[e] = (bf16_t)Wf2[e];
    }
    for (int e = tid; e < 512 * 64; e += 256) sWout[e] = (bf16_t)Wout[e];
    for (int e = tid; e < 16 * 64; e += 256)
        sWdecB[e] = (e < 512) ? (bf16_t)Wdec[e] : (bf16_t)0.0f;
    if (tid < 64) {
        sW0[tid]  = Win[tid * 65];
        sBin[tid] = b_in[tid];
        sBf1[tid] = bf1[tid];
        sBf2[tid] = bf2[tid];
    }
    for (int e = tid; e < 512; e += 256) sBout[e] = bout[e];
    if (tid < 8)  sBdec[tid] = bdec[tid];
    for (int e = tid; e < 640; e += 256) sWcls[e] = Wcls[e];
    if (tid < 10) sBcls[tid] = bcls[tid];

    // ---- z0 = a[:,0] @ Wini.T + bini ----
    {
        int bl = tid >> 3;
        const float* cb = coeffs + (size_t)(bb + bl) * (T_LEN - 1) * 32;
        float a0c[8];
#pragma unroll
        for (int i = 0; i < 8; ++i) a0c[i] = cb[i];
#pragma unroll
        for (int m = 0; m < 8; ++m) {
            int h = (tid & 7) * 8 + m;
            float acc = bini[h];
#pragma unroll
            for (int i = 0; i < 8; ++i) acc += a0c[i] * Wini[h * 8 + i];
            sZ[bl * 64 + h] = acc;
        }
    }
    if (tid < ROWS) {
        int c = 0;
        for (int t = 0; t < T_LEN; ++t) c += (maskp[(size_t)(bb + tid) * T_LEN + t] != 0);
        sLen[tid] = (c > 0) ? c : 1;
    }
    __syncthreads();

    const int mt  = w & 1;
    const int ntb = (w >> 1) * 16;
    const int hi8 = (lane >= 16) ? 8 : 0;

    // ---- hoist loop-invariant B fragments & biases into registers ----
    const v16bf wz0  = load_frag_b(sWz,  ntb, lane, 0);
    const v16bf wz1  = load_frag_b(sWz,  ntb, lane, 32);
    const v16bf wf10 = load_frag_b(sWf1, ntb, lane, 0);
    const v16bf wf11 = load_frag_b(sWf1, ntb, lane, 32);
    const v16bf wf20 = load_frag_b(sWf2, ntb, lane, 0);
    const v16bf wf21 = load_frag_b(sWf2, ntb, lane, 32);
    const v16bf wd0  = load_frag_b(sWdecB, 0, lane, 0);
    const v16bf wd1  = load_frag_b(sWdecB, 0, lane, 32);
    const float bin_l = sBin[ntb + (lane & 15)];
    const float w0_l  = sW0[ntb + (lane & 15)];
    const float bf1_l = sBf1[ntb + (lane & 15)];
    const float bf2_l = sBf2[ntb + (lane & 15)];
    const float bdec_l = ((lane & 15) < 8) ? sBdec[lane & 15] : 0.0f;
    float bout_l[8];
#pragma unroll
    for (int j = 0; j < 8; ++j) bout_l[j] = sBout[((w >> 1) + 4 * j) * 16 + (lane & 15)];

    // per-thread dX staging addresses (transposed LDS layout [i][row])
    const int dxRow = tid >> 3, dxI = tid & 7;
    const int dxLds = dxI * ROWS + dxRow;

    for (int s = 0; s < T_LEN - 1; ++s) {
        float ts  = times[s];
        float dtv = times[s + 1] - ts;

        // stage dX for this step into sDX[i][row]
        const size_t gdx = ((size_t)s * NBATCH + bb + dxRow) * 8 + dxI;
#if defined(HAVE_ASYNC_LDS)
        {
            float* gp = const_cast<float*>(&dXw[gdx]);
            gint_as1* gp1 = (gint_as1*)gp;
            lint_as3* lp3 = (lint_as3*)&sDX[dxLds];
            __builtin_amdgcn_global_load_async_to_lds_b32(gp1, lp3, 0, 0);
        }
#else
        sDX[dxLds] = dXw[gdx];
        if (s + 1 < T_LEN - 1)
            __builtin_prefetch(&dXw[gdx + (size_t)NBATCH * 8], 0, 1);
#endif

        // z -> bf16 (vectorized: 2x b128 load, 1x b128 store per thread)
        {
            int e = tid * 8;
            float4 za = *reinterpret_cast<const float4*>(&sZ[e]);
            float4 zb = *reinterpret_cast<const float4*>(&sZ[e + 4]);
            v8bf pk;
            pk[0] = (bf16_t)za.x; pk[1] = (bf16_t)za.y; pk[2] = (bf16_t)za.z; pk[3] = (bf16_t)za.w;
            pk[4] = (bf16_t)zb.x; pk[5] = (bf16_t)zb.y; pk[6] = (bf16_t)zb.z; pk[7] = (bf16_t)zb.w;
            *reinterpret_cast<v8bf*>(&sA[e]) = pk;
        }
        // logits on length match (rare path, VALU)
        {
            int bl = tid >> 3, oc = tid & 7;
            if (s == sLen[bl] - 1) {
#pragma unroll
                for (int m = 0; m < 2; ++m) {
                    int c = oc + 8 * m;
                    if (c < CDIM) {
                        float lg = sBcls[c];
                        for (int h = 0; h < 64; ++h) lg += sZ[bl * 64 + h] * sWcls[c * 64 + h];
                        outLog[(size_t)(bb + bl) * CDIM + c] = lg;
                    }
                }
            }
        }
#if defined(HAVE_ASYNC_LDS)
        wait_asynccnt0();     // drain per-wave ASYNCcnt before publishing barrier
#endif
        __syncthreads();

        // A fragments of z (row tile mt) -- shared by decode and G1
        v16bf a0 = load_frag_a(sA, mt * 16, lane, 0);
        v16bf a1 = load_frag_a(sA, mt * 16, lane, 32);

        // decode pred[:, s, :] with WMMA on waves 0..1 (mt == w there); scalar branch
        if (wu < 2) {
            v8f dacc = gemm_pre(a0, a1, wd0, wd1, bdec_l);
            int col = lane & 15;
            if (col < ODIM) {
                int rbase = mt * 16 + hi8;
#pragma unroll
                for (int r = 0; r < 8; ++r)
                    outPred[(size_t)(bb + rbase + r) * T_LEN * ODIM + s * ODIM + col] = dacc[r];
            }
        }
        // G1: z @ Wz.T + (b_in + t*w0)
        store_act(sB, gemm_pre(a0, a1, wz0, wz1, bin_l + ts * w0_l), mt, ntb, lane, false);
        __syncthreads();
        // G2: lipswish(y1 @ Wf1.T + bf1)
        a0 = load_frag_a(sB, mt * 16, lane, 0);
        a1 = load_frag_a(sB, mt * 16, lane, 32);
        store_act(sA, gemm_pre(a0, a1, wf10, wf11, bf1_l), mt, ntb, lane, true);
        __syncthreads();
        // G3: y2 @ Wf2.T + bf2
        a0 = load_frag_a(sA, mt * 16, lane, 0);
        a1 = load_frag_a(sA, mt * 16, lane, 32);
        store_act(sB, gemm_pre(a0, a1, wf20, wf21, bf2_l), mt, ntb, lane, false);
        __syncthreads();
        // G4: f = y3 @ Wout.T + bout, fused contraction z += dt * einsum(f, dX)
        a0 = load_frag_a(sB, mt * 16, lane, 0);
        a1 = load_frag_a(sB, mt * 16, lane, 32);
        float dxv[8];
        {
            int base = (lane & 7) * ROWS + mt * 16 + hi8;
            float4 d0 = *reinterpret_cast<const float4*>(&sDX[base]);
            float4 d1 = *reinterpret_cast<const float4*>(&sDX[base + 4]);
            dxv[0] = d0.x; dxv[1] = d0.y; dxv[2] = d0.z; dxv[3] = d0.w;
            dxv[4] = d1.x; dxv[5] = d1.y; dxv[6] = d1.z; dxv[7] = d1.w;
        }
#pragma unroll
        for (int j = 0; j < 8; ++j) {
            int ct = (w >> 1) + 4 * j;
            v16bf b0 = load_frag_b(sWout, ct * 16, lane, 0);
            v16bf b1 = load_frag_b(sWout, ct * 16, lane, 32);
            v8f acc = gemm_pre(a0, a1, b0, b1, bout_l[j]);
            int h = ct * 2 + ((lane & 15) >> 3);   // i = lane&7 is constant per lane
#pragma unroll
            for (int r = 0; r < 8; ++r) {
                float p = acc[r] * dxv[r];
                p += __shfl_xor(p, 1, 8);
                p += __shfl_xor(p, 2, 8);
                p += __shfl_xor(p, 4, 8);
                if ((lane & 7) == 0)
                    atomicAdd(&sZ[(mt * 16 + r + hi8) * 64 + h], dtv * p);
            }
        }
        __syncthreads();
    }

    // ---- final step: convert, logits (len==T), decode pred[:, T-1, :] ----
    {
        int e = tid * 8;
        float4 za = *reinterpret_cast<const float4*>(&sZ[e]);
        float4 zb = *reinterpret_cast<const float4*>(&sZ[e + 4]);
        v8bf pk;
        pk[0] = (bf16_t)za.x; pk[1] = (bf16_t)za.y; pk[2] = (bf16_t)za.z; pk[3] = (bf16_t)za.w;
        pk[4] = (bf16_t)zb.x; pk[5] = (bf16_t)zb.y; pk[6] = (bf16_t)zb.z; pk[7] = (bf16_t)zb.w;
        *reinterpret_cast<v8bf*>(&sA[e]) = pk;
    }
    {
        int bl = tid >> 3, oc = tid & 7;
        if (T_LEN - 1 == sLen[bl] - 1) {
#pragma unroll
            for (int m = 0; m < 2; ++m) {
                int c = oc + 8 * m;
                if (c < CDIM) {
                    float lg = sBcls[c];
                    for (int h = 0; h < 64; ++h) lg += sZ[bl * 64 + h] * sWcls[c * 64 + h];
                    outLog[(size_t)(bb + bl) * CDIM + c] = lg;
                }
            }
        }
    }
    __syncthreads();
    if (wu < 2) {
        v16bf a0 = load_frag_a(sA, mt * 16, lane, 0);
        v16bf a1 = load_frag_a(sA, mt * 16, lane, 32);
        v8f dacc = gemm_pre(a0, a1, wd0, wd1, bdec_l);
        int col = lane & 15;
        if (col < ODIM) {
            int rbase = mt * 16 + hi8;
#pragma unroll
            for (int r = 0; r < 8; ++r)
                outPred[(size_t)(bb + rbase + r) * T_LEN * ODIM + (T_LEN - 1) * ODIM + col] = dacc[r];
        }
    }
}

extern "C" void kernel_launch(void* const* d_in, const int* in_sizes, int n_in,
                              void* d_out, int out_size, void* d_ws, size_t ws_size,
                              hipStream_t stream) {
    const float* coeffs = (const float*)d_in[0];
    const float* times  = (const float*)d_in[1];
    const int*   maskp  = (const int*)  d_in[2];
    const float* Win    = (const float*)d_in[3];
    const float* b_in   = (const float*)d_in[4];
    const float* Wf1    = (const float*)d_in[5];
    const float* bf1    = (const float*)d_in[6];
    const float* Wf2    = (const float*)d_in[7];
    const float* bf2    = (const float*)d_in[8];
    const float* Wout   = (const float*)d_in[9];
    const float* bout   = (const float*)d_in[10];
    const float* Wini   = (const float*)d_in[11];
    const float* bini   = (const float*)d_in[12];
    const float* Wdec   = (const float*)d_in[13];
    const float* bdec   = (const float*)d_in[14];
    const float* Wcls   = (const float*)d_in[15];
    const float* bcls   = (const float*)d_in[16];

    float* dXw     = (float*)d_ws;                       // (T-1) * 256 * 8 f32 = 8.4 MB
    float* outPred = (float*)d_out;                      // (256, 1024, 8)
    float* outLog  = outPred + (size_t)NBATCH * T_LEN * ODIM;  // (256, 10)

    const int total = (T_LEN - 1) * NBATCH * IDIM;
    cde_dx_kernel<<<(total + 255) / 256, 256, 0, stream>>>(coeffs, times, dXw);

    cde_scan_kernel<<<NWG, 256, 0, stream>>>(
        coeffs, times, maskp, Win, b_in, Wf1, bf1, Wf2, bf2, Wout, bout,
        Wini, bini, Wdec, bdec, Wcls, bcls, dXw, outPred, outLog);
}
